// SarathiModel_8959301779829
// MI455X (gfx1250) — matrix operand order repaired and checked by reading, not compiled, not run on hardware
//
#include <hip/hip_runtime.h>
#include <hip/hip_bf16.h>
#include <math.h>

// ---------------------------------------------------------------------------
// MoE grouped MLP for gfx1250 (MI455X).
// fc2 = (gelu(X @ W1)) @ W2, grouped over 8 balanced experts.
// fp32 inputs decomposed into bf16 hi/lo planes; 3-term bf16 emulation
// (Ah*Bh + Ah*Bl + Al*Bh) on v_wmma_f32_16x16x32_bf16, f32 accumulate.
// Pipelined staging: A tiles via async global->LDS copies (double-buffered
// LDS, ASYNCcnt) when available; B tiles (transposed) via register
// double-buffering. Falls back to register double-buffering for A too.
// ---------------------------------------------------------------------------

typedef __attribute__((ext_vector_type(16))) __bf16 v16bf;
typedef __attribute__((ext_vector_type(8)))  float  v8f;
typedef __attribute__((__vector_size__(4 * sizeof(int)))) int v4i_raw;

#define E_   8
#define H_   1024
#define F_   4096
#define T_   2048
#define NTOK (E_ * T_)

#define BM      128
#define BN      128
#define BK      32
#define LDT     40          // padded LDS stride (bf16 elems) along K: 80B rows
#define THREADS 256         // 8 wave32 waves

#ifndef __has_builtin
#define __has_builtin(x) 0
#endif

#if defined(__AMDGCN__) && __has_builtin(__builtin_amdgcn_global_load_async_to_lds_b128)
#define HAVE_ASYNC 1
#else
#define HAVE_ASYNC 0
#endif

#if HAVE_ASYNC
#if __has_builtin(__builtin_amdgcn_s_wait_asynccnt)
#define ASYNC_WAIT() __builtin_amdgcn_s_wait_asynccnt(0)
#else
#define ASYNC_WAIT() asm volatile("s_wait_asynccnt 0x0" ::: "memory")
#endif
#define ASYNC_COPY_B128(gp, lp)                                              \
  __builtin_amdgcn_global_load_async_to_lds_b128(                            \
      (v4i_raw*)(gp), (v4i_raw*)(lp), 0, 0)
#endif

__device__ __forceinline__ unsigned short f2bf(float f) {
  __bf16 b = (__bf16)f;
  return __builtin_bit_cast(unsigned short, b);
}
__device__ __forceinline__ float bf2f(unsigned short u) {
  __bf16 b = __builtin_bit_cast(__bf16, u);
  return (float)b;
}

struct Q2 { uint4 a, b; };
__device__ __forceinline__ v16bf load_frag(const unsigned short* p0,
                                           const unsigned short* p1) {
  Q2 q;
  q.a = *(const uint4*)p0;   // ds_load_b128
  q.b = *(const uint4*)p1;   // ds_load_b128
  return __builtin_bit_cast(v16bf, q);
}

// ---------------------------------------------------------------------------
// Split fp32 -> bf16 hi + bf16 residual lo (RNE both steps).
// ---------------------------------------------------------------------------
__global__ __launch_bounds__(256)
void split_bf16_kernel(const float* __restrict__ src,
                       unsigned short* __restrict__ hi,
                       unsigned short* __restrict__ lo,
                       long long n4) {
  long long i = (long long)blockIdx.x * blockDim.x + threadIdx.x;
  if (i >= n4) return;
  float4 v = ((const float4*)src)[i];
  ushort4 h, l;
  h.x = f2bf(v.x); l.x = f2bf(v.x - bf2f(h.x));
  h.y = f2bf(v.y); l.y = f2bf(v.y - bf2f(h.y));
  h.z = f2bf(v.z); l.z = f2bf(v.z - bf2f(h.z));
  h.w = f2bf(v.w); l.w = f2bf(v.w - bf2f(h.w));
  ((ushort4*)hi)[i] = h;
  ((ushort4*)lo)[i] = l;
}

// ---------------------------------------------------------------------------
// Grouped GEMM core. A: [NTOK, KDIM] row-major hi/lo planes. B: [E, KDIM,
// NDIM] row-major hi/lo planes. Block = 128x128 tile for one expert; 8 waves
// in 4(M) x 2(N); per wave 2x4 frags of 16x16, 3 WMMAs per frag per k-step.
// ---------------------------------------------------------------------------
template<int KDIM, int NDIM, bool DO_GELU>
__global__ __launch_bounds__(THREADS)
void moe_gemm_kernel(const unsigned short* __restrict__ Ah,
                     const unsigned short* __restrict__ Al,
                     const unsigned short* __restrict__ Bh,
                     const unsigned short* __restrict__ Bl,
                     unsigned short* __restrict__ Ch,
                     unsigned short* __restrict__ Cl,
                     float* __restrict__ Cf) {
#if HAVE_ASYNC
  __shared__ unsigned short sAh[2][BM * LDT];   // double-buffered (async)
  __shared__ unsigned short sAl[2][BM * LDT];
#else
  __shared__ unsigned short sAh[1][BM * LDT];
  __shared__ unsigned short sAl[1][BM * LDT];
#endif
  __shared__ unsigned short sBh[BN * LDT];
  __shared__ unsigned short sBl[BN * LDT];

  constexpr int NK = KDIM / BK;
  const int e    = blockIdx.z;
  const int n0   = blockIdx.x * BN;
  const int m0   = blockIdx.y * BM;
  const int tid  = threadIdx.x;
  const int wave = tid >> 5;
  const int lane = tid & 31;
  const int wm    = wave & 3;    // wave's M sub-block (x32 rows)
  const int wn    = wave >> 2;   // wave's N sub-block (x64 cols)
  const int lhalf = lane >> 4;   // 0 | 1 (lane half)
  const int l16   = lane & 15;

  const unsigned short* gAh = Ah + ((size_t)e * T_ + m0) * KDIM;
  const unsigned short* gAl = Al + ((size_t)e * T_ + m0) * KDIM;
  const unsigned short* gBh = Bh + (size_t)e * KDIM * NDIM;
  const unsigned short* gBl = Bl + (size_t)e * KDIM * NDIM;

  // per-thread staging slots (2 x uint4 per plane per tile)
  int mA[2], kcA[2], krB[2], ncB[2];
#pragma unroll
  for (int it = 0; it < 2; ++it) {
    int idA = tid + it * THREADS;
    mA[it]  = idA >> 2;
    kcA[it] = (idA & 3) << 3;
    int idB = tid + it * THREADS;
    krB[it] = idB >> 4;
    ncB[it] = (idB & 15) << 3;
  }

  uint4 bregh[2], bregl[2];
#if !HAVE_ASYNC
  uint4 aregh[2], aregl[2];
#endif

  // ---- staging helpers ----------------------------------------------------
  auto stageA_issue = [&](int k0, int buf) {
#pragma unroll
    for (int it = 0; it < 2; ++it) {
      size_t go = (size_t)mA[it] * KDIM + k0 + kcA[it];
      int    lo = mA[it] * LDT + kcA[it];
#if HAVE_ASYNC
      ASYNC_COPY_B128(gAh + go, &sAh[buf][lo]);
      ASYNC_COPY_B128(gAl + go, &sAl[buf][lo]);
#else
      (void)buf;
      aregh[it] = *(const uint4*)(gAh + go);
      aregl[it] = *(const uint4*)(gAl + go);
#endif
    }
  };
#if !HAVE_ASYNC
  auto stageA_commit = [&]() {
#pragma unroll
    for (int it = 0; it < 2; ++it) {
      int lo = mA[it] * LDT + kcA[it];
      *(uint4*)&sAh[0][lo] = aregh[it];
      *(uint4*)&sAl[0][lo] = aregl[it];
    }
  };
#endif
  auto loadB_regs = [&](int k0) {
#pragma unroll
    for (int it = 0; it < 2; ++it) {
      size_t go = (size_t)(k0 + krB[it]) * NDIM + n0 + ncB[it];
      bregh[it] = *(const uint4*)(gBh + go);
      bregl[it] = *(const uint4*)(gBl + go);
    }
  };
  auto storeB_regs = [&]() {
#pragma unroll
    for (int it = 0; it < 2; ++it) {
      unsigned short th[8], tl[8];
      *(uint4*)th = bregh[it];
      *(uint4*)tl = bregl[it];
#pragma unroll
      for (int j = 0; j < 8; ++j) {
        sBh[(ncB[it] + j) * LDT + krB[it]] = th[j];
        sBl[(ncB[it] + j) * LDT + krB[it]] = tl[j];
      }
    }
  };

  v8f acc[2][4];
  const v8f vzero = {0.f, 0.f, 0.f, 0.f, 0.f, 0.f, 0.f, 0.f};
#pragma unroll
  for (int s = 0; s < 2; ++s)
#pragma unroll
    for (int f = 0; f < 4; ++f) acc[s][f] = vzero;

  // ---- prologue: stage tile 0 --------------------------------------------
  stageA_issue(0, 0);
  loadB_regs(0);
#if !HAVE_ASYNC
  stageA_commit();
#endif
  storeB_regs();
#if HAVE_ASYNC
  ASYNC_WAIT();
#endif
  __syncthreads();

  const int kbA = lhalf * 8;
  const int kbB = lhalf * 8;

  for (int kt = 0; kt < NK; ++kt) {
    const bool more = (kt + 1) < NK;
    if (more) {
      // stage tile kt+1 while computing tile kt
      stageA_issue((kt + 1) * BK, (kt + 1) & 1);
      loadB_regs((kt + 1) * BK);
      // L2 hint two tiles ahead (global_prefetch_b8)
      if (kt + 2 < NK) {
        __builtin_prefetch(gBh + (size_t)((kt + 2) * BK + krB[0]) * NDIM + n0 + ncB[0], 0, 0);
        __builtin_prefetch(gAh + (size_t)mA[0] * KDIM + (kt + 2) * BK + kcA[0], 0, 0);
      }
    }

    // ---- per-lane WMMA operand fetch (ISA 16-bit A/B layouts) ------------
    const unsigned short* cAh = sAh[HAVE_ASYNC ? (kt & 1) : 0];
    const unsigned short* cAl = sAl[HAVE_ASYNC ? (kt & 1) : 0];
    v16bf ah[2], al[2], bh[4], bl[4];
#pragma unroll
    for (int s = 0; s < 2; ++s) {
      int r = (wm * 32 + s * 16 + l16) * LDT;
      ah[s] = load_frag(&cAh[r + kbA], &cAh[r + kbA + 16]);
      al[s] = load_frag(&cAl[r + kbA], &cAl[r + kbA + 16]);
    }
#pragma unroll
    for (int f = 0; f < 4; ++f) {
      int c = (wn * 64 + f * 16 + l16) * LDT;
      bh[f] = load_frag(&sBh[c + kbB], &sBh[c + kbB + 16]);
      bl[f] = load_frag(&sBl[c + kbB], &sBl[c + kbB + 16]);
    }

    // ---- 3-term bf16 emulation: Ah*Bh + Ah*Bl + Al*Bh --------------------
#pragma unroll
    for (int s = 0; s < 2; ++s)
#pragma unroll
      for (int f = 0; f < 4; ++f) {
        acc[s][f] = __builtin_amdgcn_wmma_f32_16x16x32_bf16(
            false, ah[s], false, bh[f], (short)0, acc[s][f], false, false);
        acc[s][f] = __builtin_amdgcn_wmma_f32_16x16x32_bf16(
            false, ah[s], false, bl[f], (short)0, acc[s][f], false, false);
        acc[s][f] = __builtin_amdgcn_wmma_f32_16x16x32_bf16(
            false, al[s], false, bh[f], (short)0, acc[s][f], false, false);
      }

    __syncthreads();   // all waves done reading sB (and, fallback, sA)
    if (more) {
#if !HAVE_ASYNC
      stageA_commit();
#endif
      storeB_regs();
#if HAVE_ASYNC
      ASYNC_WAIT();    // tile kt+1 async copies have landed in sA[(kt+1)&1]
#endif
    }
    __syncthreads();   // staged tile kt+1 visible to all waves
  }

  // ---- epilogue; C frag layout: VGPR r -> M = r + 8*lhalf, N = l16 --------
#pragma unroll
  for (int s = 0; s < 2; ++s)
#pragma unroll
    for (int f = 0; f < 4; ++f)
#pragma unroll
      for (int r = 0; r < 8; ++r) {
        int row = m0 + wm * 32 + s * 16 + lhalf * 8 + r;
        int col = n0 + wn * 64 + f * 16 + l16;
        float v = acc[s][f][r];
        size_t idx = ((size_t)e * T_ + row) * NDIM + col;
        if constexpr (DO_GELU) {
          float g = 0.5f * v * (1.0f + erff(v * 0.70710678118654752f));
          unsigned short h = f2bf(g);
          unsigned short l = f2bf(g - bf2f(h));
          Ch[idx] = h;
          Cl[idx] = l;
        } else {
          Cf[idx] = v;
        }
      }
}

// ---------------------------------------------------------------------------
extern "C" void kernel_launch(void* const* d_in, const int* in_sizes, int n_in,
                              void* d_out, int out_size, void* d_ws,
                              size_t ws_size, hipStream_t stream) {
  const float* x  = (const float*)d_in[0];   // [NTOK, H]
  const float* w1 = (const float*)d_in[1];   // [E, H, F]
  const float* w2 = (const float*)d_in[2];   // [E, F, H]
  // d_in[3] tokens_per_expert: balanced (== T_), layout already permuted.
  float* out = (float*)d_out;

  const size_t nX  = (size_t)NTOK * H_;
  const size_t nW1 = (size_t)E_ * H_ * F_;
  const size_t nW2 = (size_t)E_ * F_ * H_;
  const size_t nA  = (size_t)NTOK * F_;

  char* ws = (char*)d_ws;
  unsigned short* xh   = (unsigned short*)ws;  ws += nX  * 2;
  unsigned short* xl   = (unsigned short*)ws;  ws += nX  * 2;
  unsigned short* w1h  = (unsigned short*)ws;  ws += nW1 * 2;
  unsigned short* w1l  = (unsigned short*)ws;  ws += nW1 * 2;
  unsigned short* w2h  = (unsigned short*)ws;  ws += nW2 * 2;
  unsigned short* w2l  = (unsigned short*)ws;  ws += nW2 * 2;
  unsigned short* acth = (unsigned short*)ws;  ws += nA  * 2;
  unsigned short* actl = (unsigned short*)ws;

  // 1) fp32 -> bf16 hi/lo planes
  split_bf16_kernel<<<(unsigned)(nX  / 4 / 256), 256, 0, stream>>>(x,  xh,  xl,  (long long)(nX  / 4));
  split_bf16_kernel<<<(unsigned)(nW1 / 4 / 256), 256, 0, stream>>>(w1, w1h, w1l, (long long)(nW1 / 4));
  split_bf16_kernel<<<(unsigned)(nW2 / 4 / 256), 256, 0, stream>>>(w2, w2h, w2l, (long long)(nW2 / 4));

  // 2) fc1 + exact GELU -> hi/lo activation planes
  moe_gemm_kernel<H_, F_, true><<<dim3(F_ / BN, T_ / BM, E_), THREADS, 0, stream>>>(
      xh, xl, w1h, w1l, acth, actl, nullptr);

  // 3) fc2 -> fp32 output
  moe_gemm_kernel<F_, H_, false><<<dim3(H_ / BN, T_ / BM, E_), THREADS, 0, stream>>>(
      acth, actl, w2h, w2l, nullptr, nullptr, out);
}
